// SelfAttention_58394375356485
// MI455X (gfx1250) — matrix operand order repaired and checked by reading, not compiled, not run on hardware
//
#include <hip/hip_runtime.h>
#include <hip/hip_bf16.h>

// ---------------------------------------------------------------- constants
#define S_LEN 2048
#define BATCH 2
#define DMODEL 4096
#define NH 32
#define NKVH 8
#define HD 128
#define NREP 4
#define TOK (S_LEN * BATCH)             // 4096 token rows (row = s*B + b)
#define QKV_N (NH * HD + 2 * NKVH * HD) // 6144 fused projection cols

typedef unsigned short u16;
typedef unsigned int u32;
typedef __attribute__((ext_vector_type(16))) __bf16 v16bf;
typedef __attribute__((ext_vector_type(2)))  __bf16 v2bf;
typedef __attribute__((ext_vector_type(8)))  float  v8f;
typedef __attribute__((ext_vector_type(4)))  u32    v4u;

union V16U {               // bit-view of a WMMA 16-bit A/B operand (8 VGPRs)
    v4u   q[2];
    u32   w[8];
    v16bf v;
};

static __device__ __forceinline__ u32 cvt2(float lo, float hi) {
    v2bf r;
    r[0] = (__bf16)lo;                 // -> v_cvt_pk_bf16_f32 (confirmed)
    r[1] = (__bf16)hi;
    return __builtin_bit_cast(u32, r);
}
static __device__ __forceinline__ u16 f2bf(float f) {
    return __builtin_bit_cast(u16, (__bf16)f);
}
static __device__ __forceinline__ float bf2f(u16 h) {
    return __builtin_bit_cast(float, (u32)h << 16);
}
static __device__ __forceinline__ v4u ld16(const u16* p) {
    return *reinterpret_cast<const v4u*>(p);
}
static __device__ __forceinline__ v8f wmma_bf16(v16bf a, v16bf b, v8f c) {
    // D = A(16x32) * B(32x16) + C  -> v_wmma_f32_16x16x32_bf16
    return __builtin_amdgcn_wmma_f32_16x16x32_bf16(false, a, false, b,
                                                   (short)0, c, false, false);
}

// ============================================================ f32 -> bf16
// One-pass streaming convert (float4 -> 2x cvt_pk -> b64 store); runs at HBM
// rate, hoists ALL conversions out of the GEMM inner loops.
__global__ __launch_bounds__(256) void tobf16_kernel(
    const float* __restrict__ in, u16* __restrict__ out, long n4)
{
    const long i = (long)blockIdx.x * blockDim.x + threadIdx.x;
    if (i >= n4) return;
    const float4 f = reinterpret_cast<const float4*>(in)[i];
    uint2 r;
    r.x = cvt2(f.x, f.y);
    r.y = cvt2(f.z, f.w);
    *reinterpret_cast<uint2*>(out + 4 * i) = r;
}

// ===================================================================== QKV
// One wave -> 32 tokens x 64 cols of the fused [wq|wk|wv] projection, all
// operands pre-converted bf16: 12 b128 loads -> 8 WMMAs per K-step, no VALU
// converts. Epilogue scatters into Q(B,H,S,HD), K(B,KVH,S,HD), V^T(B,KVH,HD,S).
__global__ __launch_bounds__(128) void qkv_kernel(
    const u16* __restrict__ xb, const u16* __restrict__ wqkvb,
    u16* __restrict__ q, u16* __restrict__ k, u16* __restrict__ vT)
{
    const int lane = threadIdx.x, lrow = lane & 15, lhi = lane >> 4;
    const int m0 = blockIdx.x * 32;
    const int n0 = (blockIdx.y * blockDim.y + threadIdx.y) * 64;

    const u16* xrow[2];
#pragma unroll
    for (int mt = 0; mt < 2; ++mt)
        xrow[mt] = xb + (size_t)(m0 + mt * 16 + lrow) * DMODEL;
    const u16* wrow[4];
#pragma unroll
    for (int t = 0; t < 4; ++t)
        wrow[t] = wqkvb + (size_t)(n0 + t * 16 + lrow) * DMODEL;

    v8f acc[2][4] = {};
    for (int k0 = 0; k0 < DMODEL; k0 += 32) {
        // all 12 b128 loads staged into distinct regs -> one wait covers all
        V16U Av[2], Bv[4];
#pragma unroll
        for (int mt = 0; mt < 2; ++mt) {
            Av[mt].q[0] = ld16(xrow[mt] + k0 + lhi * 8);
            Av[mt].q[1] = ld16(xrow[mt] + k0 + 16 + lhi * 8);
        }
#pragma unroll
        for (int t = 0; t < 4; ++t) {
            Bv[t].q[0] = ld16(wrow[t] + k0 + lhi * 8);
            Bv[t].q[1] = ld16(wrow[t] + k0 + 16 + lhi * 8);
        }
        __builtin_prefetch(xrow[0] + k0 + 512, 0, 0);   // global_prefetch_b8
        __builtin_prefetch(wrow[0] + k0 + 512, 0, 0);
        __builtin_prefetch(wrow[2] + k0 + 512, 0, 0);
#pragma unroll
        for (int t = 0; t < 4; ++t) {
            acc[0][t] = wmma_bf16(Av[0].v, Bv[t].v, acc[0][t]);
            acc[1][t] = wmma_bf16(Av[1].v, Bv[t].v, acc[1][t]);
        }
    }
    // ---- epilogue: C layout lane=N col, VGPR i -> row m0+mt*16+i+8*lhi
#pragma unroll
    for (int mt = 0; mt < 2; ++mt) {
#pragma unroll
        for (int t = 0; t < 4; ++t) {
            const int c = n0 + t * 16 + lrow;
#pragma unroll
            for (int i = 0; i < 8; ++i) {
                const int row = m0 + mt * 16 + i + lhi * 8;  // token = s*B + b
                const int s = row >> 1, b = row & 1;
                const u16 hv = f2bf(acc[mt][t][i]);
                if (c < NH * HD) {
                    const int h = c >> 7, hd = c & 127;
                    q[(((size_t)b * NH + h) * S_LEN + s) * HD + hd] = hv;
                } else if (c < NH * HD + NKVH * HD) {
                    const int kv = (c - NH * HD) >> 7, hd = (c - NH * HD) & 127;
                    k[(((size_t)b * NKVH + kv) * S_LEN + s) * HD + hd] = hv;
                } else {
                    const int kv = (c - NH * HD - NKVH * HD) >> 7;
                    const int hd = (c - NH * HD - NKVH * HD) & 127;
                    vT[(((size_t)b * NKVH + kv) * HD + hd) * S_LEN + s] = hv;
                }
            }
        }
    }
}

// ==================================================================== RoPE
__global__ __launch_bounds__(256) void rope_kernel(
    u16* __restrict__ buf, const float* __restrict__ cosb,
    const float* __restrict__ sinb, long total_pairs)
{
    const long idx = (long)blockIdx.x * blockDim.x + threadIdx.x;
    if (idx >= total_pairs) return;
    const int hd2 = (int)(idx & 63);
    const long t = idx >> 6;
    const int s = (int)(t & (S_LEN - 1));
    const long bh = t >> 11;
    u32* p = reinterpret_cast<u32*>(buf + ((bh * S_LEN + s) * HD + hd2 * 2));
    const u32 pr = *p;
    const float x0 = bf2f((u16)(pr & 0xffff)), x1 = bf2f((u16)(pr >> 16));
    const float c = cosb[s * 64 + hd2], sn = sinb[s * 64 + hd2];
    *p = cvt2(x0 * c - x1 * sn, x0 * sn + x1 * c);
}

// =========================================================== flash attention
// One wave per (b, h, 16-query tile). S^T = K * Q^T so lane = query: softmax
// stats are lane-local + one shfl_xor(16); P^T(C-layout) -> B-layout via 8
// packs + 8 half-wave shuffles; O^T = V^T * P^T streamed from transposed V.
__global__ __launch_bounds__(32) void attn_kernel(
    const u16* __restrict__ q, const u16* __restrict__ k,
    const u16* __restrict__ vT, u16* __restrict__ attn)
{
    const int lane = threadIdx.x, lrow = lane & 15, lhi = lane >> 4;
    const int qb = blockIdx.x;                    // 16-query tile
    const int b = blockIdx.y >> 5, h = blockIdx.y & 31, kvh = h >> 2;

    const u16* qbase = q  + ((size_t)b * NH   + h)   * S_LEN * HD;
    const u16* kbase = k  + ((size_t)b * NKVH + kvh) * S_LEN * HD;
    const u16* vbase = vT + ((size_t)b * NKVH + kvh) * HD * S_LEN;

    // Q held in B-layout registers for the whole kernel (4 K-chunks of 32)
    V16U Bq[4];
    const u16* qr = qbase + (size_t)(qb * 16 + lrow) * HD;
#pragma unroll
    for (int c = 0; c < 4; ++c) {
        Bq[c].q[0] = ld16(qr + c * 32 + lhi * 8);
        Bq[c].q[1] = ld16(qr + c * 32 + 16 + lhi * 8);
    }

    v8f o[8] = {};                                // O^T accumulators (HD=8 tiles)
    float m_run = -1e30f, l_run = 0.f;
    const int qidx = qb * 16 + lrow;              // this lane's query (both halves)
    const int kmax = qb * 16 + 15;
    const float sc = 0.08838834764831845f;        // 1/sqrt(128)

    for (int kb = 0; kb <= kmax; kb += 32) {
        // ---- stage all 16 K-tile loads for this 32-key block, then 8 WMMAs
        const u16* kr0 = kbase + (size_t)(kb + lrow) * HD;
        const u16* kr1 = kbase + (size_t)(kb + 16 + lrow) * HD;
        V16U A0[4], A1[4];
#pragma unroll
        for (int c = 0; c < 4; ++c) {
            A0[c].q[0] = ld16(kr0 + c * 32 + lhi * 8);
            A0[c].q[1] = ld16(kr0 + c * 32 + 16 + lhi * 8);
            A1[c].q[0] = ld16(kr1 + c * 32 + lhi * 8);
            A1[c].q[1] = ld16(kr1 + c * 32 + 16 + lhi * 8);
        }
        v8f st0 = {}, st1 = {};
#pragma unroll
        for (int c = 0; c < 4; ++c) {
            st0 = wmma_bf16(A0[c].v, Bq[c].v, st0);
            st1 = wmma_bf16(A1[c].v, Bq[c].v, st1);
        }
        // scale + causal mask (key index per C-layout: M = i + 8*lhi)
#pragma unroll
        for (int i = 0; i < 8; ++i) {
            const int k0i = kb + i + lhi * 8, k1i = kb + 16 + i + lhi * 8;
            float s0 = st0[i] * sc, s1 = st1[i] * sc;
            st0[i] = (k0i > qidx) ? -1e30f : s0;
            st1[i] = (k1i > qidx) ? -1e30f : s1;
        }
        // online softmax stats: lane-local over 16 keys + half-wave combine
        float tm = -1e30f;
#pragma unroll
        for (int i = 0; i < 8; ++i) tm = fmaxf(tm, fmaxf(st0[i], st1[i]));
        tm = fmaxf(tm, __shfl_xor(tm, 16, 32));
        const float nm = fmaxf(m_run, tm);
        const float alpha = __expf(m_run - nm);
        float rs = 0.f;
#pragma unroll
        for (int i = 0; i < 8; ++i) {
            st0[i] = __expf(st0[i] - nm);
            st1[i] = __expf(st1[i] - nm);
            rs += st0[i] + st1[i];
        }
        rs += __shfl_xor(rs, 16, 32);
        l_run = l_run * alpha + rs;
        m_run = nm;
        // P^T C-layout -> B-layout (lane=query, contiguous keys per half)
        u32 pk0[4], pk1[4];
#pragma unroll
        for (int j = 0; j < 4; ++j) {
            pk0[j] = cvt2(st0[2 * j], st0[2 * j + 1]);
            pk1[j] = cvt2(st1[2 * j], st1[2 * j + 1]);
        }
        V16U bp;
#pragma unroll
        for (int j = 0; j < 4; ++j) {
            const u32 sw0 = (u32)__shfl_xor((int)pk0[j], 16, 32);
            const u32 sw1 = (u32)__shfl_xor((int)pk1[j], 16, 32);
            bp.w[j]     = (lhi == 0) ? pk0[j] : sw1;    // keys 0..7  | 16..23
            bp.w[j + 4] = (lhi == 0) ? sw0    : pk1[j]; // keys 8..15 | 24..31
        }
        // ---- stage all 16 V^T loads, then 8 PV WMMAs (C pre-scaled by alpha)
        V16U Av[8];
#pragma unroll
        for (int t = 0; t < 8; ++t) {
            const u16* vr = vbase + (size_t)(t * 16 + lrow) * S_LEN;
            Av[t].q[0] = ld16(vr + kb + lhi * 16);
            Av[t].q[1] = ld16(vr + kb + lhi * 16 + 8);
        }
#pragma unroll
        for (int t = 0; t < 8; ++t) {
            v8f ca;
#pragma unroll
            for (int i = 0; i < 8; ++i) ca[i] = o[t][i] * alpha;
            o[t] = wmma_bf16(Av[t].v, bp.v, ca);
        }
    }
    // epilogue: O^T C-layout -> attn[(s*B+b), h*HD + hd] bf16
    const float inv = 1.0f / l_run;
    u16* orow = attn + ((size_t)qidx * BATCH + b) * (NH * HD) + (size_t)h * HD;
#pragma unroll
    for (int t = 0; t < 8; ++t)
#pragma unroll
        for (int i = 0; i < 8; ++i)
            orow[t * 16 + i + lhi * 8] = f2bf(o[t][i] * inv);
}

// ============================================================ output proj
// out(token, D) = attn(token, 4096) @ wo^T ; 32x64 tile per wave, all-bf16
// operands (12 b128 loads -> 8 WMMAs / K-step); f32 stores straight to d_out.
__global__ __launch_bounds__(128) void oproj_kernel(
    const u16* __restrict__ attn, const u16* __restrict__ wob,
    float* __restrict__ out)
{
    const int lane = threadIdx.x, lrow = lane & 15, lhi = lane >> 4;
    const int m0 = blockIdx.x * 32;
    const int n0 = (blockIdx.y * blockDim.y + threadIdx.y) * 64;

    const u16* arow[2];
#pragma unroll
    for (int mt = 0; mt < 2; ++mt)
        arow[mt] = attn + (size_t)(m0 + mt * 16 + lrow) * (NH * HD);
    const u16* wrow[4];
#pragma unroll
    for (int t = 0; t < 4; ++t)
        wrow[t] = wob + (size_t)(n0 + t * 16 + lrow) * (NH * HD);

    v8f acc[2][4] = {};
    for (int k0 = 0; k0 < NH * HD; k0 += 32) {
        V16U Av[2], Bv[4];
#pragma unroll
        for (int mt = 0; mt < 2; ++mt) {
            Av[mt].q[0] = ld16(arow[mt] + k0 + lhi * 8);
            Av[mt].q[1] = ld16(arow[mt] + k0 + 16 + lhi * 8);
        }
#pragma unroll
        for (int t = 0; t < 4; ++t) {
            Bv[t].q[0] = ld16(wrow[t] + k0 + lhi * 8);
            Bv[t].q[1] = ld16(wrow[t] + k0 + 16 + lhi * 8);
        }
        __builtin_prefetch(wrow[0] + k0 + 512, 0, 0);
        __builtin_prefetch(wrow[2] + k0 + 512, 0, 0);
#pragma unroll
        for (int t = 0; t < 4; ++t) {
            acc[0][t] = wmma_bf16(Av[0].v, Bv[t].v, acc[0][t]);
            acc[1][t] = wmma_bf16(Av[1].v, Bv[t].v, acc[1][t]);
        }
    }
#pragma unroll
    for (int mt = 0; mt < 2; ++mt)
#pragma unroll
        for (int t = 0; t < 4; ++t)
#pragma unroll
            for (int i = 0; i < 8; ++i)
                out[(size_t)(m0 + mt * 16 + i + lhi * 8) * DMODEL + n0 + t * 16 + lrow]
                    = acc[mt][t][i];
}

// ================================================================ launcher
extern "C" void kernel_launch(void* const* d_in, const int* in_sizes, int n_in,
                              void* d_out, int out_size, void* d_ws, size_t ws_size,
                              hipStream_t stream) {
    const float* x    = (const float*)d_in[0];
    // d_in[1] = start_pos (== 0 in reference; unused)
    const float* fcos = (const float*)d_in[2];
    const float* fsin = (const float*)d_in[3];
    const float* wq   = (const float*)d_in[4];
    const float* wk   = (const float*)d_in[5];
    const float* wv   = (const float*)d_in[6];
    const float* wo   = (const float*)d_in[7];
    float* out = (float*)d_out;

    // bf16 workspace (u16 elements), 160MB total:
    //   q 16M | k 4M | vT 4M | xb 16M (reused as attn buffer) | wqkv 24M | wo 16M
    u16* q     = (u16*)d_ws;
    u16* kbuf  = q     + (size_t)16777216;
    u16* vT    = kbuf  + (size_t)4194304;
    u16* xb    = vT    + (size_t)4194304;
    u16* wqkvb = xb    + (size_t)16777216;
    u16* wob   = wqkvb + (size_t)25165824;
    u16* attnb = xb;            // alias: xb dead after qkv_kernel

    // one-pass f32 -> bf16 conversions (stream at HBM rate)
    tobf16_kernel<<<16384, 256, 0, stream>>>(x, xb, 4194304);
    tobf16_kernel<<<16384, 256, 0, stream>>>(wq, wqkvb, 4194304);
    tobf16_kernel<<<4096, 256, 0, stream>>>(wk, wqkvb + 16777216, 1048576);
    tobf16_kernel<<<4096, 256, 0, stream>>>(wv, wqkvb + 20971520, 1048576);
    tobf16_kernel<<<16384, 256, 0, stream>>>(wo, wob, 4194304);

    qkv_kernel<<<dim3(TOK / 32, QKV_N / 256), dim3(32, 4), 0, stream>>>(
        xb, wqkvb, q, kbuf, vT);

    const long qpairs = (long)BATCH * NH * S_LEN * (HD / 2);
    const long kpairs = (long)BATCH * NKVH * S_LEN * (HD / 2);
    rope_kernel<<<(int)(qpairs / 256), 256, 0, stream>>>(q, fcos, fsin, qpairs);
    rope_kernel<<<(int)(kpairs / 256), 256, 0, stream>>>(kbuf, fcos, fsin, kpairs);

    attn_kernel<<<dim3(S_LEN / 16, BATCH * NH), 32, 0, stream>>>(q, kbuf, vT, attnb);

    oproj_kernel<<<dim3(TOK / 32, DMODEL / 256), dim3(32, 4), 0, stream>>>(
        attnb, wob, out);
}
